// AttentionLayer_49778670961063
// MI455X (gfx1250) — matrix operand order repaired and checked by reading, not compiled, not run on hardware
//
#include <hip/hip_runtime.h>
#include <math.h>

typedef __bf16 bf16_t;
typedef __attribute__((ext_vector_type(16))) __bf16 v16bf;
typedef __attribute__((ext_vector_type(8)))  float  v8f;

static __device__ inline v8f vzero() {
    v8f z = {0.f, 0.f, 0.f, 0.f, 0.f, 0.f, 0.f, 0.f};
    return z;
}

static __device__ inline v8f wmma_bf16(v16bf a, v16bf b, v8f c) {
    // D = A(16x32 bf16) * B(32x16 bf16) + C(16x16 f32)
    return __builtin_amdgcn_wmma_f32_16x16x32_bf16(false, a, false, b, (short)0, c, false, false);
}

// Async memory->LDS 16-byte copy (CDNA5 GLOBAL_LOAD_ASYNC_TO_LDS_B128,
// tracked by ASYNCcnt). lds_off = LDS byte offset (low 32 bits of the
// generic pointer to a __shared__ object == LDS aperture offset).
static __device__ inline void async_ld_b128(unsigned lds_off, const void* gaddr) {
    asm volatile("global_load_async_to_lds_b128 %0, %1, off"
                 :: "v"(lds_off), "v"(gaddr) : "memory");
}
static __device__ inline void wait_async() {
    asm volatile("s_wait_asynccnt 0x0" ::: "memory");
}
static __device__ inline unsigned lds_off32(const void* p) {
    return (unsigned)(uintptr_t)p;
}

// ---------------------------------------------------------------------------
// f32 -> bf16 conversion (vectorized)
// ---------------------------------------------------------------------------
__global__ __launch_bounds__(256) void cvt_bf16_k(const float* __restrict__ in,
                                                  bf16_t* __restrict__ out, int n4) {
    int i = blockIdx.x * blockDim.x + threadIdx.x;
    if (i < n4) {
        float4 v = ((const float4*)in)[i];
        union { bf16_t o[4]; uint2 u; } p;
        p.o[0] = (bf16_t)v.x; p.o[1] = (bf16_t)v.y;
        p.o[2] = (bf16_t)v.z; p.o[3] = (bf16_t)v.w;
        *(uint2*)(out + (size_t)i * 4) = p.u;
    }
}

// ---------------------------------------------------------------------------
// W (R x C) f32  ->  W^T (C x R) bf16   (32x32 LDS tiles)
// ---------------------------------------------------------------------------
__global__ __launch_bounds__(256) void transpose_k(const float* __restrict__ in,
                                                   bf16_t* __restrict__ out, int R, int C) {
    __shared__ bf16_t tile[32][33];
    int c0 = blockIdx.x * 32, r0 = blockIdx.y * 32;
    for (int i = threadIdx.y; i < 32; i += 8)
        tile[i][threadIdx.x] = (bf16_t)in[(size_t)(r0 + i) * C + c0 + threadIdx.x];
    __syncthreads();
    for (int i = threadIdx.y; i < 32; i += 8)
        out[(size_t)(c0 + i) * R + r0 + threadIdx.x] = tile[threadIdx.x][i];
}

// ---------------------------------------------------------------------------
// Generic bf16 WMMA GEMM:  C[M,N] = A[M,K] @ W[K,N]
//   A row-major bf16, WT = W^T (N x K) row-major bf16.
//   Block tile 128x128, K-step 32, 8 waves (2 x 4), double-buffered LDS.
//   LDS tiles staged in WMMA fragment order via async b128 loads (ASYNCcnt).
//   EPI: 0 = f32 store, 1 = bf16 store, 2 = kv scatter (k row-major, v^T)
// ---------------------------------------------------------------------------
template <int EPI>
__global__ __launch_bounds__(256) void gemm_bf16_k(
    const bf16_t* __restrict__ A, const bf16_t* __restrict__ WT,
    float* __restrict__ Cf, bf16_t* __restrict__ Cb,
    bf16_t* __restrict__ Kout, bf16_t* __restrict__ VTout,
    int M, int N, int K) {
    alignas(32) __shared__ bf16_t lsA[2][4096];
    alignas(32) __shared__ bf16_t lsB[2][4096];
    const int tid = threadIdx.x, lane = tid & 31;
    const int wave = tid >> 5, wm = wave >> 2, wn = wave & 3;
    const int row0 = blockIdx.y * 128, col0 = blockIdx.x * 128;
    const int KT = K >> 5;

    auto stage = [&](int kt, int buf) {
        int k0 = kt << 5;
#pragma unroll
        for (int t = 0; t < 2; ++t) {
            int p = tid + (t << 8);
            int m = p >> 2, c = p & 3;
            async_ld_b128(
                lds_off32(&lsA[buf][(m >> 4) * 512 + ((m & 15) + ((c & 1) << 4)) * 16 + ((c >> 1) << 3)]),
                A + (size_t)(row0 + m) * K + k0 + c * 8);
            async_ld_b128(
                lds_off32(&lsB[buf][(m >> 4) * 512 + ((m & 15) + ((c & 1) << 4)) * 16 + ((c >> 1) << 3)]),
                WT + (size_t)(col0 + m) * K + k0 + c * 8);
        }
    };

    v8f acc[4][2];
#pragma unroll
    for (int i = 0; i < 4; ++i) { acc[i][0] = vzero(); acc[i][1] = vzero(); }

    stage(0, 0);
    for (int kt = 0; kt < KT; ++kt) {
        wait_async();        // my async writes (issued last iter) are done
        __syncthreads();     // everyone's are visible
        if (kt + 1 < KT) stage(kt + 1, (kt + 1) & 1);
        int buf = kt & 1;
        v16bf b0 = *(const v16bf*)&lsB[buf][(wn * 2 + 0) * 512 + lane * 16];
        v16bf b1 = *(const v16bf*)&lsB[buf][(wn * 2 + 1) * 512 + lane * 16];
#pragma unroll
        for (int mb = 0; mb < 4; ++mb) {
            v16bf a = *(const v16bf*)&lsA[buf][(wm * 4 + mb) * 512 + lane * 16];
            acc[mb][0] = wmma_bf16(a, b0, acc[mb][0]);
            acc[mb][1] = wmma_bf16(a, b1, acc[mb][1]);
        }
    }

    const int n0 = lane & 15, half = lane >> 4;
#pragma unroll
    for (int mb = 0; mb < 4; ++mb)
#pragma unroll
        for (int i = 0; i < 2; ++i) {
            int col = col0 + wn * 32 + i * 16 + n0;
#pragma unroll
            for (int r = 0; r < 8; ++r) {
                int row = row0 + wm * 64 + mb * 16 + r + half * 8;
                float v = acc[mb][i][r];
                if (EPI == 0) {
                    Cf[(size_t)row * N + col] = v;
                } else if (EPI == 1) {
                    Cb[(size_t)row * N + col] = (bf16_t)v;
                } else { // kv: cols 0..63 -> K (row-major), 64..127 -> V^T per batch
                    if (col < 64)
                        Kout[(size_t)row * 64 + col] = (bf16_t)v;
                    else
                        VTout[((size_t)(row >> 11) * 64 + (col - 64)) * 2048 + (row & 2047)] = (bf16_t)v;
                }
            }
        }
}

// ---------------------------------------------------------------------------
// Flash attention: one wave per 16-query tile per (b,h). d_head = 64.
//   S = Q K^T * 0.125, online softmax, O += P V  (all via WMMA bf16)
// ---------------------------------------------------------------------------
static __device__ inline void stP(bf16_t* base, int m, int k, float v) {
    int h = (k >> 4) & 1;
    int j = k & 15;
    base[(m + (h << 4)) * 16 + j] = (bf16_t)v;
}

__global__ __launch_bounds__(256) void flash_attn_k(
    const bf16_t* __restrict__ Q, const bf16_t* __restrict__ Kb,
    const bf16_t* __restrict__ VT, bf16_t* __restrict__ Ao) {
    alignas(32) __shared__ bf16_t lsP[8][512];
    const int lane = threadIdx.x & 31, wave = threadIdx.x >> 5;
    const int t = blockIdx.x * 8 + wave;       // 0..4095
    const int b = t >> 11;                     // t / (16*128)
    const int h = (t >> 7) & 15;
    const int qt = t & 127;
    const int n0 = lane & 15, half = lane >> 4;
    const size_t qrow = (size_t)b * 2048 + qt * 16;

    // Q A-fragments (K-dim = 64 -> two 16x32 fragments), loaded once
    v16bf aq[2];
#pragma unroll
    for (int kc = 0; kc < 2; ++kc) {
        const bf16_t* qp = Q + (qrow + n0) * 1024 + h * 64 + kc * 32 + half * 16;
        union { v16bf v; uint4 u[2]; } fr;
        fr.u[0] = *(const uint4*)(qp);
        fr.u[1] = *(const uint4*)(qp + 8);
        aq[kc] = fr.v;
    }

    v8f o[4];
#pragma unroll
    for (int nb = 0; nb < 4; ++nb) o[nb] = vzero();
    float mrow[8], lrow[8];
#pragma unroll
    for (int r = 0; r < 8; ++r) { mrow[r] = -3.0e38f; lrow[r] = 0.f; }
    const float scale = 0.125f;
    bf16_t* myP = lsP[wave];

    for (int c = 0; c < 64; ++c) {  // 2048 keys / 32 per chunk
        v8f s0 = vzero(), s1 = vzero();
        const bf16_t* kp0 = Kb + ((size_t)b * 2048 + c * 32 + n0) * 64 + half * 16;
        const bf16_t* kp1 = kp0 + 16 * 64;
        s0 = wmma_bf16(aq[0], *(const v16bf*)(kp0), s0);
        s0 = wmma_bf16(aq[1], *(const v16bf*)(kp0 + 32), s0);
        s1 = wmma_bf16(aq[0], *(const v16bf*)(kp1), s1);
        s1 = wmma_bf16(aq[1], *(const v16bf*)(kp1 + 32), s1);

        float alpha[8];
#pragma unroll
        for (int r = 0; r < 8; ++r) {
            float v0 = s0[r] * scale, v1 = s1[r] * scale;
            float mx = fmaxf(v0, v1);
#pragma unroll
            for (int d = 8; d >= 1; d >>= 1) mx = fmaxf(mx, __shfl_xor(mx, d));
            float mn = fmaxf(mrow[r], mx);
            float al = __expf(mrow[r] - mn);
            float p0 = __expf(v0 - mn), p1 = __expf(v1 - mn);
            float ps = p0 + p1;
#pragma unroll
            for (int d = 8; d >= 1; d >>= 1) ps += __shfl_xor(ps, d);
            lrow[r] = lrow[r] * al + ps;
            mrow[r] = mn;
            alpha[r] = al;
            int mr = r + half * 8;
            stP(myP, mr, n0, p0);        // C-layout -> A-fragment-layout bounce
            stP(myP, mr, n0 + 16, p1);
        }
        asm volatile("s_wait_dscnt 0x0" ::: "memory");
        v16bf pf = *(const v16bf*)&myP[lane * 16];
#pragma unroll
        for (int nb = 0; nb < 4; ++nb) {
#pragma unroll
            for (int r = 0; r < 8; ++r) o[nb][r] *= alpha[r];
            const bf16_t* vp = VT + ((size_t)b * 64 + nb * 16 + n0) * 2048 + c * 32 + half * 16;
            o[nb] = wmma_bf16(pf, *(const v16bf*)vp, o[nb]);
        }
    }

    float inv[8];
#pragma unroll
    for (int r = 0; r < 8; ++r) inv[r] = 1.0f / lrow[r];
#pragma unroll
    for (int nb = 0; nb < 4; ++nb)
#pragma unroll
        for (int r = 0; r < 8; ++r) {
            size_t row = qrow + r + half * 8;
            Ao[row * 1024 + h * 64 + nb * 16 + n0] = (bf16_t)(o[nb][r] * inv[r]);
        }
}

// ---------------------------------------------------------------------------
// Dual-GEMM gated FFN: f = gelu_exact(H @ Wg) * (H @ Wa), bf16 out
//   Block tile 128M x 64N, waves 4x2, two B operands from w_ff1^T halves.
// ---------------------------------------------------------------------------
__global__ __launch_bounds__(256) void dual_ff_k(const bf16_t* __restrict__ Hb,
                                                 const bf16_t* __restrict__ W1T,
                                                 bf16_t* __restrict__ F) {
    alignas(32) __shared__ bf16_t lsA[2][4096];
    alignas(32) __shared__ bf16_t lsBa[2][2048];
    alignas(32) __shared__ bf16_t lsBg[2][2048];
    const int tid = threadIdx.x, lane = tid & 31;
    const int wave = tid >> 5, wm = wave >> 1, wn = wave & 1;
    const int row0 = blockIdx.y * 128, col0 = blockIdx.x * 64;
    const int K = 1024, KT = 32;
    const bf16_t* WaT = W1T;                        // rows 0..4095  (a half)
    const bf16_t* WgT = W1T + (size_t)4096 * 1024;  // rows 4096..8191 (gate)

    auto stage = [&](int kt, int buf) {
        int k0 = kt << 5;
#pragma unroll
        for (int t = 0; t < 2; ++t) {
            int p = tid + (t << 8);
            int m = p >> 2, c = p & 3;
            async_ld_b128(
                lds_off32(&lsA[buf][(m >> 4) * 512 + ((m & 15) + ((c & 1) << 4)) * 16 + ((c >> 1) << 3)]),
                Hb + (size_t)(row0 + m) * K + k0 + c * 8);
        }
        {
            int n = tid >> 2, c = tid & 3;
            size_t idx = (n >> 4) * 512 + ((n & 15) + ((c & 1) << 4)) * 16 + ((c >> 1) << 3);
            async_ld_b128(lds_off32(&lsBa[buf][idx]), WaT + (size_t)(col0 + n) * K + k0 + c * 8);
            async_ld_b128(lds_off32(&lsBg[buf][idx]), WgT + (size_t)(col0 + n) * K + k0 + c * 8);
        }
    };

    v8f accA[2][2], accG[2][2];
#pragma unroll
    for (int i = 0; i < 2; ++i)
#pragma unroll
        for (int j = 0; j < 2; ++j) { accA[i][j] = vzero(); accG[i][j] = vzero(); }

    stage(0, 0);
    for (int kt = 0; kt < KT; ++kt) {
        wait_async();
        __syncthreads();
        if (kt + 1 < KT) stage(kt + 1, (kt + 1) & 1);
        int buf = kt & 1;
        v16bf ba0 = *(const v16bf*)&lsBa[buf][(wn * 2 + 0) * 512 + lane * 16];
        v16bf ba1 = *(const v16bf*)&lsBa[buf][(wn * 2 + 1) * 512 + lane * 16];
        v16bf bg0 = *(const v16bf*)&lsBg[buf][(wn * 2 + 0) * 512 + lane * 16];
        v16bf bg1 = *(const v16bf*)&lsBg[buf][(wn * 2 + 1) * 512 + lane * 16];
#pragma unroll
        for (int mb = 0; mb < 2; ++mb) {
            v16bf a = *(const v16bf*)&lsA[buf][(wm * 2 + mb) * 512 + lane * 16];
            accA[mb][0] = wmma_bf16(a, ba0, accA[mb][0]);
            accA[mb][1] = wmma_bf16(a, ba1, accA[mb][1]);
            accG[mb][0] = wmma_bf16(a, bg0, accG[mb][0]);
            accG[mb][1] = wmma_bf16(a, bg1, accG[mb][1]);
        }
    }

    const int n0 = lane & 15, half = lane >> 4;
#pragma unroll
    for (int mb = 0; mb < 2; ++mb)
#pragma unroll
        for (int i = 0; i < 2; ++i) {
            int col = col0 + wn * 32 + i * 16 + n0;
#pragma unroll
            for (int r = 0; r < 8; ++r) {
                int row = row0 + wm * 32 + mb * 16 + r + half * 8;
                float a = accA[mb][i][r], g = accG[mb][i][r];
                float gel = 0.5f * g * (1.0f + erff(g * 0.70710678f));
                F[(size_t)row * 4096 + col] = (bf16_t)(gel * a);
            }
        }
}

// ---------------------------------------------------------------------------
// LayerNorm kernels (one row of 1024 per block, 256 threads)
// ---------------------------------------------------------------------------
static __device__ inline float blk_sum(float v, float* sh) {
#pragma unroll
    for (int d = 16; d >= 1; d >>= 1) v += __shfl_xor(v, d);
    if ((threadIdx.x & 31) == 0) sh[threadIdx.x >> 5] = v;
    __syncthreads();
    float r = sh[0];
#pragma unroll
    for (int w = 1; w < 8; ++w) r += sh[w];
    __syncthreads();
    return r;
}

__global__ __launch_bounds__(256) void resid_ln1_k(const float* __restrict__ x,
                                                   const float* __restrict__ proj,
                                                   const float* __restrict__ g1,
                                                   const float* __restrict__ gf,
                                                   float* __restrict__ x1,
                                                   bf16_t* __restrict__ hb) {
    __shared__ float sh[8];
    size_t row = blockIdx.x;
    const float* xr = x + row * 1024;
    const float* pr = proj + row * 1024;
    float v[4], s = 0.f, s2 = 0.f;
#pragma unroll
    for (int j = 0; j < 4; ++j) {
        int col = threadIdx.x + j * 256;
        v[j] = xr[col] + pr[col];
        s += v[j]; s2 += v[j] * v[j];
    }
    s = blk_sum(s, sh); s2 = blk_sum(s2, sh);
    float mu = s * (1.f / 1024.f);
    float rs = rsqrtf(s2 * (1.f / 1024.f) - mu * mu + 1e-5f);
    float w[4], t = 0.f, t2 = 0.f;
#pragma unroll
    for (int j = 0; j < 4; ++j) {
        int col = threadIdx.x + j * 256;
        w[j] = (v[j] - mu) * rs * g1[col];
        x1[row * 1024 + col] = w[j];
        t += w[j]; t2 += w[j] * w[j];
    }
    t = blk_sum(t, sh); t2 = blk_sum(t2, sh);
    float mu2 = t * (1.f / 1024.f);
    float rs2 = rsqrtf(t2 * (1.f / 1024.f) - mu2 * mu2 + 1e-5f);
#pragma unroll
    for (int j = 0; j < 4; ++j) {
        int col = threadIdx.x + j * 256;
        hb[row * 1024 + col] = (bf16_t)((w[j] - mu2) * rs2 * gf[col]);
    }
}

__global__ __launch_bounds__(256) void final_ln_k(const float* __restrict__ x1,
                                                  const float* __restrict__ ffn,
                                                  const float* __restrict__ g2,
                                                  float* __restrict__ out) {
    __shared__ float sh[8];
    size_t row = blockIdx.x;
    const float* xr = x1 + row * 1024;
    const float* fr = ffn + row * 1024;
    float v[4], s = 0.f, s2 = 0.f;
#pragma unroll
    for (int j = 0; j < 4; ++j) {
        int col = threadIdx.x + j * 256;
        v[j] = xr[col] + fr[col];
        s += v[j]; s2 += v[j] * v[j];
    }
    s = blk_sum(s, sh); s2 = blk_sum(s2, sh);
    float mu = s * (1.f / 1024.f);
    float rs = rsqrtf(s2 * (1.f / 1024.f) - mu * mu + 1e-5f);
#pragma unroll
    for (int j = 0; j < 4; ++j) {
        int col = threadIdx.x + j * 256;
        out[row * 1024 + col] = (v[j] - mu) * rs * g2[col];
    }
}

// ---------------------------------------------------------------------------
// Launcher
// ---------------------------------------------------------------------------
extern "C" void kernel_launch(void* const* d_in, const int* in_sizes, int n_in,
                              void* d_out, int out_size, void* d_ws, size_t ws_size,
                              hipStream_t stream) {
    (void)in_sizes; (void)n_in; (void)out_size; (void)ws_size;
    const float* x      = (const float*)d_in[0];
    const float* w_q    = (const float*)d_in[1];
    const float* w_kv   = (const float*)d_in[2];
    const float* w_cat  = (const float*)d_in[3];
    const float* gamma1 = (const float*)d_in[4];
    const float* gammaf = (const float*)d_in[5];
    const float* w_ff1  = (const float*)d_in[6];
    const float* w_ff2  = (const float*)d_in[7];
    const float* gamma2 = (const float*)d_in[8];

    const int M = 4096;  // B*L

    char* base = (char*)d_ws;
    size_t off = 0;
    auto alloc = [&](size_t bytes) -> char* {
        char* p = base + off;
        off += (bytes + 255) & ~(size_t)255;
        return p;
    };
    bf16_t* xb    = (bf16_t*)alloc((size_t)M * 1024 * 2);
    bf16_t* wqT   = (bf16_t*)alloc((size_t)1024 * 1024 * 2);
    bf16_t* wkvT  = (bf16_t*)alloc((size_t)128 * 1024 * 2);
    bf16_t* wcatT = (bf16_t*)alloc((size_t)1024 * 1024 * 2);
    bf16_t* wff1T = (bf16_t*)alloc((size_t)8192 * 1024 * 2);
    bf16_t* wff2T = (bf16_t*)alloc((size_t)1024 * 4096 * 2);
    bf16_t* qb    = (bf16_t*)alloc((size_t)M * 1024 * 2);
    bf16_t* kb    = (bf16_t*)alloc((size_t)M * 64 * 2);
    bf16_t* vtb   = (bf16_t*)alloc((size_t)M * 64 * 2);
    bf16_t* attn  = (bf16_t*)alloc((size_t)M * 1024 * 2);
    float*  proj  = (float*)alloc((size_t)M * 1024 * 4);  // reused for ffn out
    float*  x1    = (float*)alloc((size_t)M * 1024 * 4);
    bf16_t* hb    = (bf16_t*)alloc((size_t)M * 1024 * 2);
    bf16_t* fb    = (bf16_t*)alloc((size_t)M * 4096 * 2);

    // 1) precision conversion + weight transposition
    cvt_bf16_k<<<(M * 1024 / 4 + 255) / 256, 256, 0, stream>>>(x, xb, M * 1024 / 4);
    transpose_k<<<dim3(32, 32),  dim3(32, 8), 0, stream>>>(w_q,   wqT,   1024, 1024);
    transpose_k<<<dim3(4, 32),   dim3(32, 8), 0, stream>>>(w_kv,  wkvT,  1024, 128);
    transpose_k<<<dim3(32, 32),  dim3(32, 8), 0, stream>>>(w_cat, wcatT, 1024, 1024);
    transpose_k<<<dim3(256, 32), dim3(32, 8), 0, stream>>>(w_ff1, wff1T, 1024, 8192);
    transpose_k<<<dim3(32, 128), dim3(32, 8), 0, stream>>>(w_ff2, wff2T, 4096, 1024);

    // 2) q / kv projections
    gemm_bf16_k<1><<<dim3(8, 32), 256, 0, stream>>>(xb, wqT, nullptr, qb, nullptr, nullptr, M, 1024, 1024);
    gemm_bf16_k<2><<<dim3(1, 32), 256, 0, stream>>>(xb, wkvT, nullptr, nullptr, kb, vtb, M, 128, 1024);

    // 3) flash attention
    flash_attn_k<<<512, 256, 0, stream>>>(qb, kb, vtb, attn);

    // 4) output projection + residual/dual LN
    gemm_bf16_k<0><<<dim3(8, 32), 256, 0, stream>>>(attn, wcatT, proj, nullptr, nullptr, nullptr, M, 1024, 1024);
    resid_ln1_k<<<M, 256, 0, stream>>>(x, proj, gamma1, gammaf, x1, hb);

    // 5) gated FFN (dual GEMM + exact gelu) and down projection
    dual_ff_k<<<dim3(64, 32), 256, 0, stream>>>(hb, wff1T, fb);
    gemm_bf16_k<0><<<dim3(8, 32), 256, 0, stream>>>(fb, wff2T, proj, nullptr, nullptr, nullptr, M, 1024, 4096);

    // 6) final residual + LN -> d_out (f32)
    final_ln_k<<<M, 256, 0, stream>>>(x1, proj, gamma2, (float*)d_out);
}
